// ExpoCommSAgent_63728724738374
// MI455X (gfx1250) — compile-verified, matrix operand order checked by reading, not run on hardware
//
#include <hip/hip_runtime.h>
#include <hip/hip_bf16.h>

// ---------------------------------------------------------------------------
// ExpoCommSAgent forward for MI455X (gfx1250), wave32 + WMMA f16->f32.
// v3: compile-time K + full unroll -> fresh SSA fragments per k-step (no
// register-rotation movs, no phi copies, scheduler-driven pipelining),
// f16 weights pre-converted once (L2-resident).
// ---------------------------------------------------------------------------

#define BS 256
#define N_AGENTS 64
#define TOPK 8
#define IN_DIM 128
#define HID 256
#define ATT 64
#define N_ACT 16
#define ROWS (BS * N_AGENTS)   // 16384

typedef __attribute__((ext_vector_type(16))) _Float16 v16h;
typedef __attribute__((ext_vector_type(8)))  _Float16 v8h;
typedef __attribute__((ext_vector_type(4)))  _Float16 v4h;
typedef __attribute__((ext_vector_type(8)))  float    v8f;

__device__ __forceinline__ v8f wmma16(v16h a, v16h b, v8f c) {
  return __builtin_amdgcn_wmma_f32_16x16x32_f16(
      /*neg_a=*/false, a, /*neg_b=*/false, b,
      /*c_mod=*/(short)0, c, /*reuse_a=*/false, /*reuse_b=*/false);
}

// --- 16x32 A-fragment loaders (ISA 7.12.2: lane&15 = row, lane>=16 => K+8).
// p points at (row, k0 + ((lane>>4)<<3)); needs elements [0..7] and [16..23].
__device__ __forceinline__ v16h frag_load(const float* __restrict__ p) {
  const float4 a0 = *(const float4*)(p + 0);
  const float4 a1 = *(const float4*)(p + 4);
  const float4 a2 = *(const float4*)(p + 16);
  const float4 a3 = *(const float4*)(p + 20);
  v16h f;
  f[0]  = (_Float16)a0.x; f[1]  = (_Float16)a0.y; f[2]  = (_Float16)a0.z; f[3]  = (_Float16)a0.w;
  f[4]  = (_Float16)a1.x; f[5]  = (_Float16)a1.y; f[6]  = (_Float16)a1.z; f[7]  = (_Float16)a1.w;
  f[8]  = (_Float16)a2.x; f[9]  = (_Float16)a2.y; f[10] = (_Float16)a2.z; f[11] = (_Float16)a2.w;
  f[12] = (_Float16)a3.x; f[13] = (_Float16)a3.y; f[14] = (_Float16)a3.z; f[15] = (_Float16)a3.w;
  return f;
}
__device__ __forceinline__ v16h frag_load(const _Float16* __restrict__ p) {
  const v8h a = *(const v8h*)(p + 0);
  const v8h b = *(const v8h*)(p + 16);
  v16h f;
#pragma unroll
  for (int i = 0; i < 8; ++i) { f[i] = a[i]; f[8 + i] = b[i]; }
  return f;
}

// ---------------------------------------------------------------------------
// Weight pre-conversion f32 -> f16 (one pass; result lives in L2 for the run).
// ---------------------------------------------------------------------------
__global__ __launch_bounds__(256)
void cvt_f32_f16(const float* __restrict__ src, _Float16* __restrict__ dst, int n4) {
  const int i = blockIdx.x * 256 + threadIdx.x;
  if (i < n4) {
    const float4 v = ((const float4*)src)[i];
    v4h h;
    h[0] = (_Float16)v.x; h[1] = (_Float16)v.y;
    h[2] = (_Float16)v.z; h[3] = (_Float16)v.w;
    ((v4h*)dst)[i] = h;
  }
}

// ---------------------------------------------------------------------------
// Generic WMMA GEMM:  out[M,N] = act( A[M,K] @ Wh[N,K]^T + bias ), Wh in f16.
// Block = 256 threads = 8 waves; each wave: 16 rows x 64 cols (4 accums).
// K is compile-time: fully unrolled k-loop, fragments are fresh SSA values
// per step (no copies); scheduler hoists global_load_b128 across WMMAs.
// ---------------------------------------------------------------------------
template <typename AT, int K, int RELU>
__global__ __launch_bounds__(256)
void wmma_gemm(const AT* __restrict__ A, int lda,
               const _Float16* __restrict__ Wh, int ldw,
               const float* __restrict__ bias,
               float* __restrict__ outF, int ldcF,
               _Float16* __restrict__ outH, int ldcH,
               int M, int N) {
  const int lane = threadIdx.x & 31;
  const int wave = threadIdx.x >> 5;
  const int row0 = blockIdx.x * 128 + wave * 16;
  const int n0   = blockIdx.y * 64;
  if (row0 >= M) return;

  const int r  = row0 + (lane & 15);
  const int kb = (lane >> 4) << 3;
  const AT* arow = A + (size_t)r * lda + kb;

  const _Float16* wrow[4];
#pragma unroll
  for (int t = 0; t < 4; ++t) {
    int nrow = n0 + t * 16 + (lane & 15);
    nrow = nrow < N ? nrow : N - 1;              // clamp for N<64 strips
    wrow[t] = Wh + (size_t)nrow * ldw + kb;
  }

  v8f acc[4] = {v8f{}, v8f{}, v8f{}, v8f{}};

#pragma unroll
  for (int k0 = 0; k0 < K; k0 += 32) {
    const v16h af = frag_load(arow + k0);
#pragma unroll
    for (int t = 0; t < 4; ++t)
      acc[t] = wmma16(af, frag_load(wrow[t] + k0), acc[t]);
  }

  // C layout (ISA 7.12.2): VGPR j -> row j (lanes 0-15) / row 8+j (lanes 16-31)
  const int rbase = row0 + ((lane >> 4) << 3);
#pragma unroll
  for (int t = 0; t < 4; ++t) {
    const int col = n0 + t * 16 + (lane & 15);
    if (col >= N) continue;
    const float bv = bias[col];
#pragma unroll
    for (int j = 0; j < 8; ++j) {
      float v = acc[t][j] + bv;
      if (RELU) v = fmaxf(v, 0.0f);
      const int rr = rbase + j;
      if (outF) outF[(size_t)rr * ldcF + col] = v;
      if (outH) outH[(size_t)rr * ldcH + col] = (_Float16)v;
    }
  }
}

// ---------------------------------------------------------------------------
// Fused GRU: 16 rows per block; each wave owns 16 hidden columns and computes
// gi/gh for its (r,z,n) gate triplet (6 WMMA accumulators), then applies the
// gate math entirely in registers.  Block = 512 threads (16 waves = 256 cols).
// Fully unrolled (8 k-steps x 6 WMMA).  Writes h (f32 out) + h (f16 concat).
// ---------------------------------------------------------------------------
__global__ __launch_bounds__(512)
void gru_fused(const _Float16* __restrict__ X,    // [ROWS, HID] f16 (relu(fc1))
               const float* __restrict__ Hprev,   // [ROWS, HID] f32
               const _Float16* __restrict__ Wih,  // [3*HID, HID] f16
               const float* __restrict__ bih,
               const _Float16* __restrict__ Whh,  // [3*HID, HID] f16
               const float* __restrict__ bhh,
               float* __restrict__ Hout,          // [ROWS, HID] -> d_out h
               _Float16* __restrict__ CatH) {     // [ROWS, 2*HID], cols 0..255
  const int lane = threadIdx.x & 31;
  const int wave = threadIdx.x >> 5;       // 0..15
  const int row0 = blockIdx.x * 16;
  const int c0   = wave * 16;              // column tile within HID
  const int r    = row0 + (lane & 15);
  const int kb   = (lane >> 4) << 3;

  const _Float16* xrow = X + (size_t)r * HID + kb;
  const float*    hrow = Hprev + (size_t)r * HID + kb;
  const _Float16* wi[3];
  const _Float16* wh[3];
#pragma unroll
  for (int g = 0; g < 3; ++g) {
    const int wr = g * HID + c0 + (lane & 15);
    wi[g] = Wih + (size_t)wr * HID + kb;
    wh[g] = Whh + (size_t)wr * HID + kb;
  }

  v8f gi[3] = {v8f{}, v8f{}, v8f{}};
  v8f gh[3] = {v8f{}, v8f{}, v8f{}};

#pragma unroll
  for (int k0 = 0; k0 < HID; k0 += 32) {
    const v16h ax = frag_load(xrow + k0);
    const v16h ah = frag_load(hrow + k0);
#pragma unroll
    for (int g = 0; g < 3; ++g) {
      gi[g] = wmma16(ax, frag_load(wi[g] + k0), gi[g]);
      gh[g] = wmma16(ah, frag_load(wh[g] + k0), gh[g]);
    }
  }

  const int col   = c0 + (lane & 15);
  const int rbase = row0 + ((lane >> 4) << 3);
  const float bir = bih[col],            bhr = bhh[col];
  const float biz = bih[HID + col],      bhz = bhh[HID + col];
  const float bin2 = bih[2 * HID + col], bhn2 = bhh[2 * HID + col];
#pragma unroll
  for (int j = 0; j < 8; ++j) {
    const int rr = rbase + j;
    const float i_r = gi[0][j] + bir,  h_r = gh[0][j] + bhr;
    const float i_z = gi[1][j] + biz,  h_z = gh[1][j] + bhz;
    const float i_n = gi[2][j] + bin2, h_n = gh[2][j] + bhn2;
    const float rg = 1.0f / (1.0f + __expf(-(i_r + h_r)));
    const float zg = 1.0f / (1.0f + __expf(-(i_z + h_z)));
    const float ng = tanhf(i_n + rg * h_n);
    const float hp = Hprev[(size_t)rr * HID + col];
    const float h  = (1.0f - zg) * ng + zg * hp;
    Hout[(size_t)rr * HID + col]       = h;
    CatH[(size_t)rr * (2 * HID) + col] = (_Float16)h;
  }
}

// ---------------------------------------------------------------------------
// Attention over gathered top-k neighbors: one wave32 per agent.
// logits via lane-split dot + shfl_xor tree, softmax in registers,
// weighted sum over pre-projected V rows (hits L2: 64 rows reused per batch).
// ---------------------------------------------------------------------------
__global__ __launch_bounds__(256)
void attn_agg(const float* __restrict__ Q,     // [ROWS, ATT]
              const float* __restrict__ Kall,  // [ROWS, ATT]
              const float* __restrict__ Vall,  // [ROWS, HID]
              const int* __restrict__ topk,    // [ROWS, TOPK]
              float* __restrict__ MaggF,       // d_out m_agg [ROWS, HID]
              _Float16* __restrict__ CatH) {   // concat buf, cols 256..511
  const int lane  = threadIdx.x & 31;
  const int agent = blockIdx.x * 8 + (threadIdx.x >> 5);
  const int bbase = (agent >> 6) << 6;         // b * N_AGENTS

  const float q0 = Q[(size_t)agent * ATT + 2 * lane];
  const float q1 = Q[(size_t)agent * ATT + 2 * lane + 1];

  int rows[TOPK];
  float lg[TOPK];
#pragma unroll
  for (int j = 0; j < TOPK; ++j) {
    const int row = bbase + topk[agent * TOPK + j];
    rows[j] = row;
    const float* kp = Kall + (size_t)row * ATT;
    float p = q0 * kp[2 * lane] + q1 * kp[2 * lane + 1];
#pragma unroll
    for (int m = 16; m >= 1; m >>= 1) p += __shfl_xor(p, m, 32);
    lg[j] = p * 0.125f;                        // 1/sqrt(ATT)
  }
  float mx = lg[0];
#pragma unroll
  for (int j = 1; j < TOPK; ++j) mx = fmaxf(mx, lg[j]);
  float e[TOPK], s = 0.0f;
#pragma unroll
  for (int j = 0; j < TOPK; ++j) { e[j] = __expf(lg[j] - mx); s += e[j]; }
  const float inv = 1.0f / s;

  float acc[8] = {0, 0, 0, 0, 0, 0, 0, 0};
#pragma unroll
  for (int j = 0; j < TOPK; ++j) {
    const float w = e[j] * inv;
    const float* vp = Vall + (size_t)rows[j] * HID;
#pragma unroll
    for (int t = 0; t < 8; ++t) acc[t] += w * vp[lane + 32 * t];
  }
#pragma unroll
  for (int t = 0; t < 8; ++t) {
    const int col = lane + 32 * t;
    MaggF[(size_t)agent * HID + col] = acc[t];
    CatH[(size_t)agent * (2 * HID) + HID + col] = (_Float16)acc[t];
  }
}

// ---------------------------------------------------------------------------
extern "C" void kernel_launch(void* const* d_in, const int* in_sizes, int n_in,
                              void* d_out, int out_size, void* d_ws, size_t ws_size,
                              hipStream_t stream) {
  const float* inputs  = (const float*)d_in[0];
  const float* hidden  = (const float*)d_in[1];
  const float* msg     = (const float*)d_in[2];
  const int*   topk    = (const int*)  d_in[3];
  const float* fc1_w   = (const float*)d_in[4];
  const float* fc1_b   = (const float*)d_in[5];
  const float* w_ih    = (const float*)d_in[6];
  const float* b_ih    = (const float*)d_in[7];
  const float* w_hh    = (const float*)d_in[8];
  const float* b_hh    = (const float*)d_in[9];
  const float* q_w     = (const float*)d_in[10];
  const float* q_b     = (const float*)d_in[11];
  const float* k_w     = (const float*)d_in[12];
  const float* k_b     = (const float*)d_in[13];
  const float* v_w     = (const float*)d_in[14];
  const float* v_b     = (const float*)d_in[15];
  const float* fc2a_w  = (const float*)d_in[16];
  const float* fc2a_b  = (const float*)d_in[17];
  const float* fc2b_w  = (const float*)d_in[18];
  const float* fc2b_b  = (const float*)d_in[19];

  // Output layout: q_out [ROWS,16] | h [ROWS,256] | m_agg [ROWS,256]
  float* q_out = (float*)d_out;
  float* h_out = q_out + (size_t)ROWS * N_ACT;
  float* m_out = h_out + (size_t)ROWS * HID;

  // Workspace carve-up
  char* ws = (char*)d_ws;
  _Float16* Xh    = (_Float16*)ws;  ws += (size_t)ROWS * HID * 2;      // relu(fc1), f16
  _Float16* CatH  = (_Float16*)ws;  ws += (size_t)ROWS * 2 * HID * 2;  // [h | m_agg] f16
  float*    Kall  = (float*)ws;     ws += (size_t)ROWS * ATT * 4;
  float*    Vall  = (float*)ws;     ws += (size_t)ROWS * HID * 4;
  float*    Qbuf  = (float*)ws;     ws += (size_t)ROWS * ATT * 4;
  _Float16* Hid2h = (_Float16*)ws;  ws += (size_t)ROWS * HID * 2;
  // f16 weight copies (converted once; live in the 192MB L2 thereafter)
  _Float16* fc1_wh  = (_Float16*)ws; ws += (size_t)HID * IN_DIM * 2;
  _Float16* w_ih_h  = (_Float16*)ws; ws += (size_t)3 * HID * HID * 2;
  _Float16* w_hh_h  = (_Float16*)ws; ws += (size_t)3 * HID * HID * 2;
  _Float16* q_wh    = (_Float16*)ws; ws += (size_t)ATT * HID * 2;
  _Float16* k_wh    = (_Float16*)ws; ws += (size_t)ATT * HID * 2;
  _Float16* v_wh    = (_Float16*)ws; ws += (size_t)HID * HID * 2;
  _Float16* fc2a_wh = (_Float16*)ws; ws += (size_t)HID * 2 * HID * 2;
  _Float16* fc2b_wh = (_Float16*)ws; ws += (size_t)N_ACT * HID * 2;
  (void)ws_size; (void)n_in; (void)in_sizes; (void)out_size;

  // 0) one-shot weight conversion to f16
  struct { const float* s; _Float16* d; int n; } cv[8] = {
    {fc1_w,  fc1_wh,  HID * IN_DIM},
    {w_ih,   w_ih_h,  3 * HID * HID},
    {w_hh,   w_hh_h,  3 * HID * HID},
    {q_w,    q_wh,    ATT * HID},
    {k_w,    k_wh,    ATT * HID},
    {v_w,    v_wh,    HID * HID},
    {fc2a_w, fc2a_wh, HID * 2 * HID},
    {fc2b_w, fc2b_wh, N_ACT * HID},
  };
  for (int i = 0; i < 8; ++i) {
    const int n4 = cv[i].n / 4;
    cvt_f32_f16<<<dim3((n4 + 255) / 256), dim3(256), 0, stream>>>(cv[i].s, cv[i].d, n4);
  }

  const dim3 blk(256);
  // 1) X = relu(inputs @ fc1^T + b)  -> f16
  wmma_gemm<float, IN_DIM, 1><<<dim3(ROWS / 128, HID / 64), blk, 0, stream>>>(
      inputs, IN_DIM, fc1_wh, IN_DIM, fc1_b,
      nullptr, 0, Xh, HID, ROWS, HID);
  // 2) Kall = msg @ k_w^T + b ;  Vall = msg @ v_w^T + b   (hoisted projections)
  wmma_gemm<float, HID, 0><<<dim3(ROWS / 128, 1), blk, 0, stream>>>(
      msg, HID, k_wh, HID, k_b,
      Kall, ATT, nullptr, 0, ROWS, ATT);
  wmma_gemm<float, HID, 0><<<dim3(ROWS / 128, HID / 64), blk, 0, stream>>>(
      msg, HID, v_wh, HID, v_b,
      Vall, HID, nullptr, 0, ROWS, HID);
  // 3) Fused GRU -> h (f32 output) + h (f16 into concat buffer)
  gru_fused<<<dim3(ROWS / 16), dim3(512), 0, stream>>>(
      Xh, hidden, w_ih_h, b_ih, w_hh_h, b_hh, h_out, CatH);
  // 4) Q = h @ q_w^T + b
  wmma_gemm<_Float16, HID, 0><<<dim3(ROWS / 128, 1), blk, 0, stream>>>(
      CatH, 2 * HID, q_wh, HID, q_b,
      Qbuf, ATT, nullptr, 0, ROWS, ATT);
  // 5) Gather + attention + aggregate -> m_agg (f32 out) + concat f16
  attn_agg<<<dim3(ROWS / 8), blk, 0, stream>>>(Qbuf, Kall, Vall, topk, m_out, CatH);
  // 6) hid2 = relu([h|m_agg] @ fc2a^T + b)  (K = 512)
  wmma_gemm<_Float16, 2 * HID, 1><<<dim3(ROWS / 128, HID / 64), blk, 0, stream>>>(
      CatH, 2 * HID, fc2a_wh, 2 * HID, fc2a_b,
      nullptr, 0, Hid2h, HID, ROWS, HID);
  // 7) q_out = hid2 @ fc2b^T + b  (N = 16, clamped B-loads / masked stores)
  wmma_gemm<_Float16, HID, 0><<<dim3(ROWS / 128, 1), blk, 0, stream>>>(
      Hid2h, HID, fc2b_wh, HID, fc2b_b,
      q_out, N_ACT, nullptr, 0, ROWS, N_ACT);
}
